// UpBlock_6408091206412
// MI455X (gfx1250) — matrix operand order repaired and checked by reading, not compiled, not run on hardware
//
#include <hip/hip_runtime.h>

// ---------------------------------------------------------------------------
// CDNA5 (gfx1250) sparse-conv UpBlock: gather -> bf16 WMMA GEMM -> scatter-add
// ---------------------------------------------------------------------------

typedef __attribute__((ext_vector_type(16))) __bf16 v16bf;
typedef __attribute__((ext_vector_type(8)))  __bf16 v8bf;
typedef __attribute__((ext_vector_type(8)))  float  v8f;

#define CO 64

__device__ __forceinline__ unsigned pack_bf16x2(float x, float y)
{
    union { __bf16 h[2]; unsigned u; } p;
    p.h[0] = (__bf16)x; p.h[1] = (__bf16)y;
    return p.u;
}

// One block = one kernel-offset k x (128*MT)-row tile. 8 waves; each wave owns
// MT 16-row A tiles and computes 16*MT x 64 outputs with
// v_wmma_f32_16x16x32_bf16 (B fragments reused across the MT A tiles), then
// scatter-adds f32 results with global atomics.
template<int CIN, int MT>
__global__ __launch_bounds__(256)
void spconv_wmma_kernel(const float* __restrict__ feats,   // [n_in, CIN]
                        const float* __restrict__ W,       // [K, CIN, CO]
                        const int*   __restrict__ idx_in,  // [K, M]
                        const int*   __restrict__ idx_out, // [K, M]
                        float*       __restrict__ out,     // [n_out, CO]
                        int M)
{
    constexpr int TM = 128 * MT;
    __shared__ __align__(32) __bf16 sA[TM * CIN]; // gathered rows, [r][c]
    __shared__ __align__(32) __bf16 sW[CO * CIN]; // W[k] transposed, [n][c]

    const int k     = blockIdx.y;
    const int mbase = blockIdx.x * TM;
    const int tid   = threadIdx.x;

    const int* ik = idx_in  + (size_t)k * M;
    const int* ok = idx_out + (size_t)k * M;

    // ---- Stage A: gather rows as b128, f32 -> bf16, packed b64 LDS stores ----
    for (int i = tid; i < (TM * CIN) / 4; i += 256) {
        int e = 4 * i;
        int r = e / CIN, c = e % CIN;
        int m = mbase + r;
        float4 f = make_float4(0.0f, 0.0f, 0.0f, 0.0f);
        if (m < M)
            f = *(const float4*)(feats + (size_t)ik[m] * CIN + c);
        union { unsigned u2[2]; unsigned long long u; } p;
        p.u2[0] = pack_bf16x2(f.x, f.y);
        p.u2[1] = pack_bf16x2(f.z, f.w);
        ((unsigned long long*)sA)[i] = p.u;
    }

    // ---- Stage W[k]: transpose [c][n] -> [n][c], f32 -> bf16 ----
    const float* Wk = W + (size_t)k * CIN * CO;
    for (int i = tid; i < (CO * CIN) / 2; i += 256) {
        int e = 2 * i;
        int n = e / CIN, c = e % CIN;
        ((unsigned*)sW)[i] =
            pack_bf16x2(Wk[(size_t)c * CO + n], Wk[(size_t)(c + 1) * CO + n]);
    }

    __syncthreads();

    const int lane = tid & 31;
    const int wv   = tid >> 5;            // wave id
    const int half = lane >> 4;           // lane group 0/1
    const int nloc = lane & 15;
    const int kb   = 8 * half;            // ISA A-fragment interleave base
    const int row0 = 16 * MT * wv + nloc; // first A row fed by this lane

    v8f acc[MT][4];
    #pragma unroll
    for (int mt = 0; mt < MT; ++mt)
        #pragma unroll
        for (int t = 0; t < 4; ++t)
            acc[mt][t] = v8f{};

    for (int c = 0; c < CIN; c += 32) {
        // A fragments: K = 16*half + (0..15) contiguous, matching B
        // (two ds_load_b128 each)
        v16bf a[MT];
        #pragma unroll
        for (int mt = 0; mt < MT; ++mt) {
            const int ar = (row0 + 16 * mt) * CIN + c + 16 * half;
            v8bf alo = *(const v8bf*)&sA[ar];
            v8bf ahi = *(const v8bf*)&sA[ar + 8];
            #pragma unroll
            for (int j = 0; j < 8; ++j) { a[mt][j] = alo[j]; a[mt][j + 8] = ahi[j]; }
        }

        #pragma unroll
        for (int t = 0; t < 4; ++t) {
            // B fragment: col = t*16 + nloc, K = 16*half + (0..15) contiguous;
            // reused from registers across the MT A tiles.
            const int col = t * 16 + nloc;
            v16bf b = *(const v16bf*)&sW[col * CIN + c + 16 * half];
            #pragma unroll
            for (int mt = 0; mt < MT; ++mt)
                acc[mt][t] = __builtin_amdgcn_wmma_f32_16x16x32_bf16(
                                 false, a[mt], false, b, (short)0, acc[mt][t],
                                 false, false);
        }
    }

    // ---- Scatter-add f32 D tiles (C/D layout: row = v + 8*half) ----
    #pragma unroll
    for (int mt = 0; mt < MT; ++mt) {
        #pragma unroll
        for (int v = 0; v < 8; ++v) {
            int m = mbase + 16 * MT * wv + 16 * mt + v + 8 * half;
            if (m < M) {
                float* o = out + (size_t)ok[m] * CO + nloc;
                atomicAdd(o +  0, acc[mt][0][v]);
                atomicAdd(o + 16, acc[mt][1][v]);
                atomicAdd(o + 32, acc[mt][2][v]);
                atomicAdd(o + 48, acc[mt][3][v]);
            }
        }
    }
}

// ---- Fused LeakyReLU (in place) + per-channel sum / sum-of-squares ----
__global__ __launch_bounds__(256)
void lrelu_stats_kernel(float* __restrict__ x, int rows, float* __restrict__ stats)
{
    __shared__ float s_sum[CO];
    __shared__ float s_sq[CO];
    const int tid = threadIdx.x;
    if (tid < CO) { s_sum[tid] = 0.0f; s_sq[tid] = 0.0f; }
    __syncthreads();

    const int total  = rows * CO;
    const int stride = gridDim.x * 256;
    float ls = 0.0f, lq = 0.0f;
    for (int i = blockIdx.x * 256 + tid; i < total; i += stride) {
        float v = x[i];
        v = v > 0.0f ? v : 0.01f * v;   // LeakyReLU(0.01)
        x[i] = v;
        ls += v; lq += v * v;
    }
    const int ch = tid & (CO - 1);      // stride is a multiple of 64
    atomicAdd(&s_sum[ch], ls);
    atomicAdd(&s_sq[ch], lq);
    __syncthreads();
    if (tid < CO) {
        atomicAdd(&stats[tid],      s_sum[tid]);
        atomicAdd(&stats[CO + tid], s_sq[tid]);
    }
}

// ---- BatchNorm apply (training-mode, biased variance) ----
__global__ __launch_bounds__(256)
void bn_apply_kernel(float* __restrict__ x, int rows,
                     const float* __restrict__ stats,
                     const float* __restrict__ g,
                     const float* __restrict__ b)
{
    const int tid0 = blockIdx.x * 256 + threadIdx.x;
    const int ch   = tid0 & (CO - 1);
    const float inv_n = 1.0f / (float)rows;
    const float mean  = stats[ch] * inv_n;
    const float var   = stats[CO + ch] * inv_n - mean * mean;
    const float scale = rsqrtf(var + 1e-5f) * g[ch];
    const float shift = b[ch] - mean * scale;

    const int total  = rows * CO;
    const int stride = gridDim.x * 256;
    for (int i = tid0; i < total; i += stride)
        x[i] = x[i] * scale + shift;
}

// ---------------------------------------------------------------------------
extern "C" void kernel_launch(void* const* d_in, const int* in_sizes, int n_in,
                              void* d_out, int out_size, void* d_ws, size_t ws_size,
                              hipStream_t stream)
{
    (void)in_sizes; (void)n_in; (void)out_size; (void)ws_size;

    constexpr int N   = 120000;
    constexpr int NUP = 240000;

    const float* x_F   = (const float*)d_in[0];
    const float* skipF = (const float*)d_in[1];
    const float* W_tr  = (const float*)d_in[2];
    const float* W_up  = (const float*)d_in[3];
    const float* W1    = (const float*)d_in[4];
    const float* W2    = (const float*)d_in[5];
    const float* W3    = (const float*)d_in[6];
    const float* g0 = (const float*)d_in[7];  const float* b0 = (const float*)d_in[8];
    const float* g1 = (const float*)d_in[9];  const float* b1 = (const float*)d_in[10];
    const float* g2 = (const float*)d_in[11]; const float* b2 = (const float*)d_in[12];
    const float* g3 = (const float*)d_in[13]; const float* b3 = (const float*)d_in[14];
    const int* mt_in = (const int*)d_in[15];  const int* mt_out = (const int*)d_in[16];
    const int* mu_in = (const int*)d_in[17];  const int* mu_out = (const int*)d_in[18];
    const int* m1_in = (const int*)d_in[19];  const int* m1_out = (const int*)d_in[20];
    const int* m2_in = (const int*)d_in[21];  const int* m2_out = (const int*)d_in[22];
    const int* m3_in = (const int*)d_in[23];  const int* m3_out = (const int*)d_in[24];

    float* bufA  = (float*)d_ws;                 // [N,   64] coarse features
    float* bufB  = bufA + (size_t)N * CO;        // [NUP, 64] ping
    float* stats = bufB + (size_t)NUP * CO;      // [128] sum | sumsq
    float* outF  = (float*)d_out;                // [NUP, 64] pong / final

    const dim3 blk(256);
    const int  EW = 2048;                        // elementwise grid
    const dim3 gTr ((N   + 127) / 128, 27);      // Cin=128, TILE_M=128
    const dim3 gUp ((NUP + 255) / 256, 27);      // Cin=64,  TILE_M=256
    const dim3 g9  ((NUP + 255) / 256,  9);
    const dim3 g27 ((NUP + 255) / 256, 27);

    // L0: trans_dilao (Cin=128) -> LeakyReLU -> BN, into bufA
    hipMemsetAsync(bufA, 0, (size_t)N * CO * sizeof(float), stream);
    spconv_wmma_kernel<128, 1><<<gTr, blk, 0, stream>>>(x_F, W_tr, mt_in, mt_out, bufA, N);
    hipMemsetAsync(stats, 0, 2 * CO * sizeof(float), stream);
    lrelu_stats_kernel<<<EW, blk, 0, stream>>>(bufA, N, stats);
    bn_apply_kernel<<<EW, blk, 0, stream>>>(bufA, N, stats, g0, b0);

    // Up-conv: init with skip (residual), scatter-accumulate on top -> bufB
    hipMemcpyAsync(bufB, skipF, (size_t)NUP * CO * sizeof(float),
                   hipMemcpyDeviceToDevice, stream);
    spconv_wmma_kernel<64, 2><<<gUp, blk, 0, stream>>>(bufA, W_up, mu_in, mu_out, bufB, NUP);

    // conv1 -> outF, lrelu+bn
    hipMemsetAsync(outF, 0, (size_t)NUP * CO * sizeof(float), stream);
    spconv_wmma_kernel<64, 2><<<g9, blk, 0, stream>>>(bufB, W1, m1_in, m1_out, outF, NUP);
    hipMemsetAsync(stats, 0, 2 * CO * sizeof(float), stream);
    lrelu_stats_kernel<<<EW, blk, 0, stream>>>(outF, NUP, stats);
    bn_apply_kernel<<<EW, blk, 0, stream>>>(outF, NUP, stats, g1, b1);

    // conv2 -> bufB, lrelu+bn
    hipMemsetAsync(bufB, 0, (size_t)NUP * CO * sizeof(float), stream);
    spconv_wmma_kernel<64, 2><<<g9, blk, 0, stream>>>(outF, W2, m2_in, m2_out, bufB, NUP);
    hipMemsetAsync(stats, 0, 2 * CO * sizeof(float), stream);
    lrelu_stats_kernel<<<EW, blk, 0, stream>>>(bufB, NUP, stats);
    bn_apply_kernel<<<EW, blk, 0, stream>>>(bufB, NUP, stats, g2, b2);

    // conv3 -> outF, lrelu+bn (final result in d_out)
    hipMemsetAsync(outF, 0, (size_t)NUP * CO * sizeof(float), stream);
    spconv_wmma_kernel<64, 2><<<g27, blk, 0, stream>>>(bufB, W3, m3_in, m3_out, outF, NUP);
    hipMemsetAsync(stats, 0, 2 * CO * sizeof(float), stream);
    lrelu_stats_kernel<<<EW, blk, 0, stream>>>(outF, NUP, stats);
    bn_apply_kernel<<<EW, blk, 0, stream>>>(outF, NUP, stats, g3, b3);
}
